// LucidRains_11759620457189
// MI455X (gfx1250) — compile-verified, matrix operand order checked by reading, not run on hardware
//
#include <hip/hip_runtime.h>

typedef __bf16 bf16;
typedef __attribute__((ext_vector_type(16))) __bf16 v16bf;
typedef __attribute__((ext_vector_type(8)))  __bf16 v8bf;
typedef __attribute__((ext_vector_type(8)))  float  v8f;

constexpr int NB = 64, BALL = 256, DIM = 512;
constexpr int H = 8, HK = 2, DH = 128, G = H / HK;       // G = 4
constexpr int W = 32, S = 16, SW = 32;
constexpr int NW = (BALL - W) / S + 1;                   // 15
constexpr int NBLK = BALL / 32;                          // 8 (BSEL=32)
constexpr int NTOK = NB * BALL;                          // 16384
constexpr int WD = W * DH;                               // 4096
constexpr float SCALE = 0.08838834764831845f;            // DH^-0.5

// ---------------------------------------------------------------------------
// WMMA helpers (CDNA5 wave32, 16x16x32 bf16 -> f32)
// A layout (16-bit, 16x32): lane L: m=L&15, half=L>>4; element e holds
//   K = e + (e&8) + 8*half. B fragment mirrors this pattern per column, so a
//   K-major ("transposed") operand gives contiguous 16-byte fragment loads.
// D layout (f32 16x16): lane L: col = L&15, element r: row = r + 8*(L>>4).
// ---------------------------------------------------------------------------
__device__ inline v16bf ldfrag(const bf16* rowp, int ks, int half) {
  const v8bf* p = (const v8bf*)(rowp + ks * 32);
  v8bf lo = p[half];
  v8bf hi = p[2 + half];
  v16bf f;
#pragma unroll
  for (int e = 0; e < 8; ++e) { f[e] = lo[e]; f[8 + e] = hi[e]; }
  return f;
}

__device__ inline v8f wmma_bf(v16bf a, v16bf b, v8f c) {
  return __builtin_amdgcn_wmma_f32_16x16x32_bf16(false, a, false, b, (short)0, c,
                                                 false, false);
}

#define LDS_PRODUCED() asm volatile("" ::: "memory")

// The Tensor Data Mover writes LDS behind the compiler's back. If an LDS
// object has no stores at all, LLVM proves loads from it are undef and folds
// the WMMA fragment reads (seen in rounds 2-3: wmma v[0:7],v[0:7], no ds
// loads). This never-executing (but not provably dead) store loop gives the
// object a potential store, so the workgroup fences in __syncthreads() force
// real ds_load_b128 fragment reads after each barrier.
__device__ inline void lds_tdm_target(bf16* p, int n) {
  if (__builtin_expect((int)blockIdx.x < 0, 0)) {
    for (int i = threadIdx.x; i < n; i += blockDim.x) p[i] = (bf16)0.f;
  }
}

// ---------------------------------------------------------------------------
// Tensor Data Mover (gfx1250): async 2-D tile DMA global -> LDS.
// Descriptor per CDNA5 ISA; data_size=2B tiles.
// ---------------------------------------------------------------------------
#if __has_builtin(__builtin_amdgcn_tensor_load_to_lds)
#define HAVE_TDM 1
typedef __attribute__((ext_vector_type(4))) unsigned int u32x4;
typedef __attribute__((ext_vector_type(8))) int i32x8;
typedef __attribute__((ext_vector_type(4))) int i32x4;

__device__ inline void tdm_load_2d(unsigned lds_off, const bf16* gptr,
                                   unsigned tile_rows, unsigned rows_avail,
                                   unsigned row_elems, unsigned stride_elems) {
  unsigned long long ga = (unsigned long long)(uintptr_t)gptr;
  u32x4 g0;
  g0[0] = 1u;                                    // count=1 (valid user D#)
  g0[1] = lds_off;                               // lds_addr (bytes)
  g0[2] = (unsigned)ga;                          // global_addr[31:0]
  g0[3] = (unsigned)((ga >> 32) & 0x01ffffffull) // global_addr[56:32]
        | (2u << 30);                            // type = 2 ("image")
  i32x8 g1;
  g1[0] = 0x1 << 16;                             // data_size = 1 -> 2 bytes
  g1[1] = (int)((stride_elems & 0xffffu) << 16); // tensor_dim0[15:0]
  g1[2] = (int)((stride_elems >> 16) & 0xffffu)  // tensor_dim0[31:16]
        | (int)((rows_avail & 0xffffu) << 16);   // tensor_dim1[15:0]
  g1[3] = (int)((rows_avail >> 16) & 0xffffu)    // tensor_dim1[31:16]
        | (int)(row_elems << 16);                // tile_dim0
  g1[4] = (int)tile_rows;                        // tile_dim1 (tile_dim2 = 0)
  g1[5] = (int)stride_elems;                     // tensor_dim0_stride[31:0]
  g1[6] = 0;                                     // stride hi / dim1_stride
  g1[7] = 0;
  i32x4 gz = {0, 0, 0, 0};
#if __clang_major__ >= 23
  i32x8 gz8 = {0, 0, 0, 0, 0, 0, 0, 0};
  __builtin_amdgcn_tensor_load_to_lds(g0, g1, gz, gz, gz8, 0);
#else
  __builtin_amdgcn_tensor_load_to_lds(g0, g1, gz, gz, 0);
#endif
}
#else
#define HAVE_TDM 0
#endif

// ---------------------------------------------------------------------------
// f32 -> bf16 with transpose: src is K x N row-major, dst is N x K (K-major)
// ---------------------------------------------------------------------------
__global__ void f2bf_t(const float* __restrict__ s, bf16* __restrict__ d,
                       int K, int N) {
  size_t i = (size_t)blockIdx.x * 256 + threadIdx.x;
  if (i >= (size_t)K * N) return;
  int k = (int)(i % K);
  int n = (int)(i / K);
  d[(size_t)n * K + k] = (bf16)s[(size_t)k * N + n];
}

// ---------------------------------------------------------------------------
// prep: x += rel @ W_pe + b_pe ; RMSNorm * gamma -> xn (f32 + bf16)
// ---------------------------------------------------------------------------
__global__ __launch_bounds__(256) void prep(const float* __restrict__ x,
                                            const float* __restrict__ pos,
                                            const float* __restrict__ wpe,
                                            const float* __restrict__ bpe,
                                            const float* __restrict__ gamma,
                                            float* __restrict__ xn,
                                            bf16* __restrict__ xnb) {
  __shared__ float sp[3][256];
  int tid = threadIdx.x, b = blockIdx.x;
  int t = b * BALL + tid;
#pragma unroll
  for (int c = 0; c < 3; ++c) sp[c][tid] = pos[(size_t)t * 3 + c];
  __syncthreads();
  for (int off = 128; off > 0; off >>= 1) {
    if (tid < off)
#pragma unroll
      for (int c = 0; c < 3; ++c) sp[c][tid] += sp[c][tid + off];
    __syncthreads();
  }
  float rel[3];
#pragma unroll
  for (int c = 0; c < 3; ++c) rel[c] = pos[(size_t)t * 3 + c] - sp[c][0] * (1.f / BALL);
  float ss = 0.f;
  for (int d = 0; d < DIM; ++d) {
    float v = x[(size_t)t * DIM + d] + rel[0] * wpe[d] + rel[1] * wpe[DIM + d] +
              rel[2] * wpe[2 * DIM + d] + bpe[d];
    xn[(size_t)t * DIM + d] = v;
    ss += v * v;
  }
  float sc = rsqrtf(ss * (1.f / DIM) + 1e-6f);
  for (int d = 0; d < DIM; ++d) {
    float v = xn[(size_t)t * DIM + d] * sc * gamma[d];
    xn[(size_t)t * DIM + d] = v;
    xnb[(size_t)t * DIM + d] = (bf16)v;
  }
}

// ---------------------------------------------------------------------------
// bf16 WMMA GEMM:  C[M,N] = act(A[M,K] @ Bt[N,K]^T)
// 128x64 block tile, BK=32, 8 waves x (32x32 register tile) = 4 WMMA/wave/step.
// TDM path: double-buffered tensor_load_to_lds pipeline (DMA of step k+1
// overlaps WMMA of step k; in-order TDM completion => s_wait_tensorcnt(2)
// guarantees the current pair landed). Fallback: vectorized uint4 copies.
// ---------------------------------------------------------------------------
__global__ __launch_bounds__(256) void gemm_bf16(const bf16* __restrict__ A,
                                                 const bf16* __restrict__ Bt,
                                                 float* __restrict__ C,
                                                 bf16* __restrict__ Cb,
                                                 int M, int N, int K, int act) {
  // two (A 128x32 | B 64x32) tile pairs: 12KB each, ping-pong
  __shared__ __align__(16) bf16 tiles[2 * (128 + 64) * 32];
  int tid = threadIdx.x;
  int wave = tid >> 5, lane = tid & 31, half = lane >> 4, l16 = lane & 15;
  int m0 = blockIdx.y * 128, n0 = blockIdx.x * 64;
  int wm = wave >> 1, wn = wave & 1;
  v8f acc[2][2] = {};

#if HAVE_TDM
  lds_tdm_target(tiles, 2 * (128 + 64) * 32);
  unsigned rA = (unsigned)((M - m0) < 128 ? (M - m0) : 128);
  unsigned rB = (unsigned)((N - n0) < 64 ? (N - n0) : 64);
  auto issue = [&](int kk, int buf) {
    tdm_load_2d((unsigned)buf * 12288u, A + (size_t)m0 * K + kk, 128u, rA, 32u,
                (unsigned)K);
    tdm_load_2d((unsigned)buf * 12288u + 8192u, Bt + (size_t)n0 * K + kk, 64u, rB,
                32u, (unsigned)K);
  };
  if (tid < 32) issue(0, 0);
  int cur = 0;
  for (int k0 = 0; k0 < K; k0 += 32) {
    if (tid < 32) {
      if (k0 + 32 < K) {
        issue(k0 + 32, cur ^ 1);
        __builtin_amdgcn_s_wait_tensorcnt(2);  // current pair complete
      } else {
        __builtin_amdgcn_s_wait_tensorcnt(0);
      }
    }
    __syncthreads();
    LDS_PRODUCED();
    const bf16* As = tiles + cur * 6144;
    const bf16* Bs = tiles + cur * 6144 + 4096;
    v16bf a0 = ldfrag(As + (wm * 32 + l16) * 32, 0, half);
    v16bf a1 = ldfrag(As + (wm * 32 + 16 + l16) * 32, 0, half);
    v16bf b0 = ldfrag(Bs + (wn * 32 + l16) * 32, 0, half);
    v16bf b1 = ldfrag(Bs + (wn * 32 + 16 + l16) * 32, 0, half);
    acc[0][0] = wmma_bf(a0, b0, acc[0][0]);
    acc[0][1] = wmma_bf(a0, b1, acc[0][1]);
    acc[1][0] = wmma_bf(a1, b0, acc[1][0]);
    acc[1][1] = wmma_bf(a1, b1, acc[1][1]);
    __syncthreads();
    cur ^= 1;
  }
#else
  bf16* As = tiles;
  bf16* Bs = tiles + 128 * 32;
  for (int k0 = 0; k0 < K; k0 += 32) {
    __syncthreads();
    {
      uint4 z4 = {0, 0, 0, 0};
      for (int c = tid; c < 512; c += 256) {  // A tile 128x32, 16B chunks
        int r = c >> 2, off = (c & 3) * 8;
        int gm = m0 + r;
        uint4 v = z4;
        if (gm < M) v = *(const uint4*)(A + (size_t)gm * K + k0 + off);
        ((uint4*)As)[c] = v;
      }
      for (int c = tid; c < 256; c += 256) {  // B tile 64x32
        int r = c >> 2, off = (c & 3) * 8;
        int gn = n0 + r;
        uint4 v = z4;
        if (gn < N) v = *(const uint4*)(Bt + (size_t)gn * K + k0 + off);
        ((uint4*)Bs)[c] = v;
      }
    }
    __syncthreads();
    v16bf a0 = ldfrag(As + (wm * 32 + l16) * 32, 0, half);
    v16bf a1 = ldfrag(As + (wm * 32 + 16 + l16) * 32, 0, half);
    v16bf b0 = ldfrag(Bs + (wn * 32 + l16) * 32, 0, half);
    v16bf b1 = ldfrag(Bs + (wn * 32 + 16 + l16) * 32, 0, half);
    acc[0][0] = wmma_bf(a0, b0, acc[0][0]);
    acc[0][1] = wmma_bf(a0, b1, acc[0][1]);
    acc[1][0] = wmma_bf(a1, b0, acc[1][0]);
    acc[1][1] = wmma_bf(a1, b1, acc[1][1]);
  }
#endif
#pragma unroll
  for (int mi = 0; mi < 2; ++mi)
#pragma unroll
    for (int ni = 0; ni < 2; ++ni) {
      int col = n0 + wn * 32 + ni * 16 + l16;
      if (col < N) {
#pragma unroll
        for (int r = 0; r < 8; ++r) {
          int row = m0 + wm * 32 + mi * 16 + r + 8 * half;
          if (row < M) {
            float vv = acc[mi][ni][r];
            if (act) vv = vv > 0.f ? vv : 0.f;
            if (C) C[(size_t)row * N + col] = vv;
            if (Cb) Cb[(size_t)row * N + col] = (bf16)vv;
          }
        }
      }
    }
}

// ---------------------------------------------------------------------------
// RoPE on q and k; bf16 copies; v stored transposed per (b,hk): vT[d][token]
// ---------------------------------------------------------------------------
__global__ __launch_bounds__(512) void rope(const float* __restrict__ q,
                                            const float* __restrict__ k,
                                            const float* __restrict__ v,
                                            float* __restrict__ qr,
                                            bf16* __restrict__ qrb,
                                            bf16* __restrict__ krb,
                                            bf16* __restrict__ vtb) {
  int t = blockIdx.x, tid = threadIdx.x;
  int n = t % BALL, b = t / BALL;
  {
    int h = tid >> 6, i = tid & 63;
    float inv = __expf(-(float)i * (9.210340371976184f / 64.f));
    float ang = (float)n * inv;
    float c = __cosf(ang), s = __sinf(ang);
    size_t bp = (size_t)t * (H * DH) + h * DH;
    float t1 = q[bp + i], t2 = q[bp + 64 + i];
    float o1 = t1 * c - t2 * s, o2 = t2 * c + t1 * s;
    qr[bp + i] = o1;        qr[bp + 64 + i] = o2;
    qrb[bp + i] = (bf16)o1; qrb[bp + 64 + i] = (bf16)o2;
  }
  if (tid < HK * 64) {
    int hk = tid >> 6, i = tid & 63;
    float inv = __expf(-(float)i * (9.210340371976184f / 64.f));
    float ang = (float)n * inv;
    float c = __cosf(ang), s = __sinf(ang);
    size_t bp = (size_t)t * (HK * DH) + hk * DH;
    float t1 = k[bp + i], t2 = k[bp + 64 + i];
    krb[bp + i] = (bf16)(t1 * c - t2 * s);
    krb[bp + 64 + i] = (bf16)(t2 * c + t1 * s);
  }
  if (tid < HK * DH) {
    int hk = tid >> 7, d = tid & (DH - 1);
    float vv = v[(size_t)t * (HK * DH) + tid];
    vtb[(((size_t)b * HK + hk) * DH + d) * BALL + n] = (bf16)vv;
  }
}

// ---------------------------------------------------------------------------
// Compression windows kw/vw = (k|v)[window] + (k|v)_pos, bf16
// ---------------------------------------------------------------------------
__global__ __launch_bounds__(256) void windows(const float* __restrict__ k,
                                               const float* __restrict__ v,
                                               const float* __restrict__ kpos,
                                               const float* __restrict__ vpos,
                                               bf16* __restrict__ kw,
                                               bf16* __restrict__ vw) {
  size_t i = (size_t)blockIdx.x * 256 + threadIdx.x;
  if (i >= (size_t)HK * NB * NW * W * DH) return;
  int d = (int)(i & (DH - 1));
  size_t r = i >> 7;
  int j = (int)(r & (W - 1)); r >>= 5;
  int w = (int)(r % NW);      r /= NW;
  int b = (int)(r % NB);
  int hk = (int)(r / NB);
  int tok = b * BALL + w * S + j;
  size_t src = (size_t)tok * (HK * DH) + hk * DH + d;
  size_t ps = (size_t)(hk * W + j) * DH + d;
  kw[i] = (bf16)(k[src] + kpos[ps]);
  vw[i] = (bf16)(v[src] + vpos[ps]);
}

// ---------------------------------------------------------------------------
// ck/cv full = [mem | compressed], f32, layout ((b*HK+hk)*16+j)*DH+d
// ---------------------------------------------------------------------------
__global__ __launch_bounds__(256) void assemble(const float* __restrict__ ckbuf,
                                                const float* __restrict__ cvbuf,
                                                const float* __restrict__ memk,
                                                const float* __restrict__ memv,
                                                float* __restrict__ ckf,
                                                float* __restrict__ cvf) {
  int i = blockIdx.x * 256 + threadIdx.x;
  int d = i & (DH - 1);
  int r = i >> 7;
  int jj = r & 15; r >>= 4;
  int hk = r % HK;
  int b = r / HK;
  float kvv, vvv;
  if (jj == 0) {
    kvv = memk[hk * DH + d];
    vvv = memv[hk * DH + d];
  } else {
    size_t src = ((size_t)hk * NB * NW + (size_t)b * NW + (jj - 1)) * DH + d;
    kvv = ckbuf[src];
    vvv = cvbuf[src];
  }
  ckf[i] = kvv;
  cvf[i] = vvv;
}

// ---------------------------------------------------------------------------
// Compressed attention + block top-1 selection. Block = (b,hk), thread = row.
// ---------------------------------------------------------------------------
__global__ __launch_bounds__(256) void cmp_attn(const float* __restrict__ qr,
                                                const float* __restrict__ ckf,
                                                const float* __restrict__ cvf,
                                                float* __restrict__ coutp,
                                                unsigned* __restrict__ selm) {
  __shared__ float cks[16 * DH];
  __shared__ float cvs[16 * DH];
  int bidx = blockIdx.x;
  int b = bidx / HK, hk = bidx % HK;
  int tid = threadIdx.x;
  {
    const float4* s0 = (const float4*)(ckf + (size_t)bidx * 16 * DH);
    const float4* s1 = (const float4*)(cvf + (size_t)bidx * 16 * DH);
    for (int i = tid; i < 16 * DH / 4; i += 256) {
      ((float4*)cks)[i] = s0[i];
      ((float4*)cvs)[i] = s1[i];
    }
  }
  __syncthreads();
  int row = tid;
  int tok = b * BALL + row;
  float imp[NW];
#pragma unroll
  for (int w = 0; w < NW; ++w) imp[w] = 0.f;

  for (int g = 0; g < G; ++g) {
    int h = hk * G + g;
    size_t qb = (size_t)tok * (H * DH) + h * DH;
    float sim[16];
#pragma unroll
    for (int j = 0; j < 16; ++j) sim[j] = 0.f;
    for (int d = 0; d < DH; ++d) {
      float qd = qr[qb + d];
#pragma unroll
      for (int j = 0; j < 16; ++j) sim[j] += qd * cks[j * DH + d];
    }
    float mx = -1e30f;
#pragma unroll
    for (int j = 0; j < 16; ++j) {
      bool vis = (j == 0) || (row >= S * (j - 1) + W - 1);
      sim[j] = vis ? sim[j] * SCALE : -1e30f;
      mx = fmaxf(mx, sim[j]);
    }
    float att[16], sum = 0.f;
#pragma unroll
    for (int j = 0; j < 16; ++j) { att[j] = __expf(sim[j] - mx); sum += att[j]; }
    float rinv = 1.f / sum;
#pragma unroll
    for (int j = 0; j < 16; ++j) att[j] *= rinv;
#pragma unroll
    for (int w = 0; w < NW; ++w) imp[w] += att[w + 1];
    for (int d = 0; d < DH; ++d) {
      float o = 0.f;
#pragma unroll
      for (int j = 0; j < 16; ++j) o += att[j] * cvs[j * DH + d];
      coutp[qb + d] = o;
    }
  }
  float bs[NBLK];
#pragma unroll
  for (int bk = 0; bk < NBLK; ++bk) bs[bk] = 0.f;
#pragma unroll
  for (int w = 0; w < NW; ++w) bs[w >> 1] += imp[w] * (1.f / G);
  int best = 0; float bv = bs[0];
#pragma unroll
  for (int bk = 1; bk < NBLK; ++bk)
    if (bs[bk] > bv) { bv = bs[bk]; best = bk; }
  selm[(size_t)bidx * BALL + row] = (1u << best) | (1u << (row >> 5));
}

// ---------------------------------------------------------------------------
// Fine-selected + sliding attention via WMMA (TDM-fed LDS tiles).
// ---------------------------------------------------------------------------
__device__ inline void build_attn_rows(const float* sim, bf16* attnb,
                                       const unsigned* selm, size_t selbase,
                                       int row0, int pass, int tid) {
  if (tid < 32) {
    int qrow = row0 + tid;
    unsigned msk = selm[selbase + qrow];
    const float* srow = sim + (size_t)tid * BALL;
    bf16* arow = attnb + (size_t)tid * BALL;
    float mx = -1e30f;
    for (int j = 0; j <= qrow; ++j) {
      bool vis = pass == 0 ? ((msk >> (j >> 5)) & 1u) : ((qrow - j) < SW);
      if (vis) mx = fmaxf(mx, srow[j]);
    }
    float sum = 0.f;
    for (int j = 0; j < BALL; ++j) {
      bool vis = (j <= qrow) &&
                 (pass == 0 ? ((msk >> (j >> 5)) & 1u) : ((qrow - j) < SW));
      float e = vis ? __expf(srow[j] - mx) : 0.f;
      sum += e;
      arow[j] = (bf16)e;
    }
    float rinv = 1.f / sum;
    for (int j = 0; j < BALL; ++j) arow[j] = (bf16)((float)arow[j] * rinv);
  }
}

__global__ __launch_bounds__(256) void fine_attn(const bf16* __restrict__ qb,
                                                 const bf16* __restrict__ kb,
                                                 const bf16* __restrict__ vtb,
                                                 const unsigned* __restrict__ selm,
                                                 float* __restrict__ foutp,
                                                 float* __restrict__ soutp) {
  __shared__ __align__(16) char smem[65536];
  float* sim = (float*)smem;            // [32][256]  32 KB          @ 0
  bf16* qs = (bf16*)(smem + 32768);     // [32][128]  (phase 1)      @ 32768
  bf16* attnb = (bf16*)(smem + 32768);  // [32][256]  (phase 2)      @ 32768
  bf16* chunk = (bf16*)(smem + 49152);  // k[64][128] / vT[128][64]  @ 49152

  int bz = blockIdx.z;
  int b = bz / HK, hk = bz % HK;
  int g = blockIdx.y;
  int rt = blockIdx.x;
  int h = hk * G + g;
  int tid = threadIdx.x;
  int wave = tid >> 5, lane = tid & 31, half = lane >> 4, l16 = lane & 15;
  int row0 = rt * 32;
  size_t selbase = (size_t)bz * BALL;

#if HAVE_TDM
  lds_tdm_target((bf16*)smem, 65536 / 2);  // TDM writes qs/chunk regions
  if (tid < 32)
    tdm_load_2d(32768u, qb + (size_t)(b * BALL + row0) * (H * DH) + h * DH, 32u,
                32u, 128u, (unsigned)(H * DH));
#else
  for (int c = tid; c < 32 * DH / 8; c += 256) {
    int r = c >> 4, off = (c & 15) * 8;
    ((uint4*)qs)[c] =
        *(const uint4*)(qb + (size_t)(b * BALL + row0 + r) * (H * DH) + h * DH + off);
  }
#endif

  // sim = SCALE * q @ k^T, 4 chunks of 64 keys
  {
    int mt = wave & 1, nt = wave >> 1;
    for (int ch = 0; ch < 4; ++ch) {
      __syncthreads();
#if HAVE_TDM
      if (tid < 32) {
        tdm_load_2d(49152u, kb + (size_t)(b * BALL + ch * 64) * (HK * DH) + hk * DH,
                    64u, 64u, 128u, (unsigned)(HK * DH));
        __builtin_amdgcn_s_wait_tensorcnt(0);
      }
#else
      for (int c = tid; c < 64 * DH / 8; c += 256) {
        int r = c >> 4, off = (c & 15) * 8;
        ((uint4*)chunk)[c] = *(const uint4*)(
            kb + (size_t)(b * BALL + ch * 64 + r) * (HK * DH) + hk * DH + off);
      }
#endif
      __syncthreads();
      LDS_PRODUCED();
      v8f acc = {};
#pragma unroll
      for (int ks = 0; ks < 4; ++ks) {
        v16bf a = ldfrag(qs + (mt * 16 + l16) * DH, ks, half);
        v16bf bf = ldfrag(chunk + (nt * 16 + l16) * DH, ks, half);
        acc = wmma_bf(a, bf, acc);
      }
      int coln = ch * 64 + nt * 16 + l16;
#pragma unroll
      for (int r = 0; r < 8; ++r)
        sim[(size_t)(mt * 16 + r + 8 * half) * BALL + coln] = acc[r] * SCALE;
    }
  }
  __syncthreads();

  for (int pass = 0; pass < 2; ++pass) {
    build_attn_rows(sim, attnb, selm, selbase, row0, pass, tid);
    __syncthreads();
    int mt = wave & 1, ntp = wave >> 1;
    v8f acc0 = {}, acc1 = {};
    for (int ch = 0; ch < 4; ++ch) {
      __syncthreads();
#if HAVE_TDM
      if (tid < 32) {
        // vT chunk: 128 rows (dims) x 64 tokens, row stride BALL
        tdm_load_2d(49152u, vtb + ((size_t)(b * HK + hk) * DH) * BALL + ch * 64,
                    128u, 128u, 64u, (unsigned)BALL);
        __builtin_amdgcn_s_wait_tensorcnt(0);
      }
#else
      for (int c = tid; c < DH * 8; c += 256) {
        int d = c >> 3, off = (c & 7) * 8;
        ((uint4*)chunk)[c] = *(const uint4*)(
            vtb + (((size_t)b * HK + hk) * DH + d) * BALL + ch * 64 + off);
      }
#endif
      __syncthreads();
      LDS_PRODUCED();
#pragma unroll
      for (int ks = 0; ks < 2; ++ks) {
        v16bf a = ldfrag(attnb + (mt * 16 + l16) * BALL + ch * 64, ks, half);
        v16bf b0 = ldfrag(chunk + (ntp * 32 + l16) * 64, ks, half);
        v16bf b1 = ldfrag(chunk + (ntp * 32 + 16 + l16) * 64, ks, half);
        acc0 = wmma_bf(a, b0, acc0);
        acc1 = wmma_bf(a, b1, acc1);
      }
    }
    float* outp = pass == 0 ? foutp : soutp;
#pragma unroll
    for (int nn = 0; nn < 2; ++nn) {
      v8f acc = nn ? acc1 : acc0;
      int col = h * DH + ntp * 32 + nn * 16 + l16;
#pragma unroll
      for (int r = 0; r < 8; ++r) {
        int qrow = row0 + mt * 16 + r + 8 * half;
        outp[(size_t)(b * BALL + qrow) * (H * DH) + col] = acc[r];
      }
    }
    __syncthreads();
  }
}

// ---------------------------------------------------------------------------
// gate + combine -> bf16 head-major (N, H*DH)
// ---------------------------------------------------------------------------
__global__ __launch_bounds__(256) void combine(const float* __restrict__ graw,
                                               const float* __restrict__ coutp,
                                               const float* __restrict__ foutp,
                                               const float* __restrict__ soutp,
                                               bf16* __restrict__ aout) {
  size_t i = (size_t)blockIdx.x * 256 + threadIdx.x;
  int col = (int)(i % (H * DH));
  size_t tok = i / (H * DH);
  int h = col / DH;
  const float* gr = graw + tok * (3 * H) + h * 3;
  float g0 = 1.f / (1.f + __expf(-gr[0]));
  float g1 = 1.f / (1.f + __expf(-gr[1]));
  float g2 = 1.f / (1.f + __expf(-gr[2]));
  aout[i] = (bf16)(g0 * coutp[i] + g1 * foutp[i] + g2 * soutp[i]);
}

// ---------------------------------------------------------------------------
// host launcher
// ---------------------------------------------------------------------------
extern "C" void kernel_launch(void* const* d_in, const int* in_sizes, int n_in,
                              void* d_out, int out_size, void* d_ws, size_t ws_size,
                              hipStream_t stream) {
  (void)in_sizes; (void)n_in; (void)out_size; (void)ws_size;
  const float* x = (const float*)d_in[0];
  const float* pos = (const float*)d_in[1];
  const float* wpe = (const float*)d_in[2];
  const float* bpe = (const float*)d_in[3];
  const float* gamma = (const float*)d_in[4];
  const float* Wq = (const float*)d_in[5];
  const float* Wk = (const float*)d_in[6];
  const float* Wv = (const float*)d_in[7];
  const float* kpos = (const float*)d_in[8];
  const float* vpos = (const float*)d_in[9];
  const float* Wck1 = (const float*)d_in[10];
  const float* Wck2 = (const float*)d_in[11];
  const float* Wcv1 = (const float*)d_in[12];
  const float* Wcv2 = (const float*)d_in[13];
  const float* memk = (const float*)d_in[14];
  const float* memv = (const float*)d_in[15];
  const float* Wg = (const float*)d_in[16];
  const float* Wo = (const float*)d_in[17];
  float* out = (float*)d_out;

  char* base = (char*)d_ws;
  size_t off = 0;
  auto alloc = [&](size_t bytes) -> char* {
    char* p = base + off;
    off = (off + bytes + 255) & ~(size_t)255;
    return p;
  };
  float* xn    = (float*)alloc((size_t)NTOK * DIM * 4);
  bf16*  xnb   = (bf16*) alloc((size_t)NTOK * DIM * 2);
  float* qf    = (float*)alloc((size_t)NTOK * H * DH * 4);
  float* kf    = (float*)alloc((size_t)NTOK * HK * DH * 4);
  float* vf    = (float*)alloc((size_t)NTOK * HK * DH * 4);
  float* graw  = (float*)alloc((size_t)NTOK * 3 * H * 4);
  float* qr    = (float*)alloc((size_t)NTOK * H * DH * 4);
  bf16*  qrb   = (bf16*) alloc((size_t)NTOK * H * DH * 2);
  bf16*  krb   = (bf16*) alloc((size_t)NTOK * HK * DH * 2);
  bf16*  vtb   = (bf16*) alloc((size_t)NTOK * HK * DH * 2);
  bf16*  kw    = (bf16*) alloc((size_t)HK * NB * NW * WD * 2);
  bf16*  vw    = (bf16*) alloc((size_t)HK * NB * NW * WD * 2);
  bf16*  hid   = (bf16*) alloc((size_t)NB * NW * WD * 2);
  float* ckbuf = (float*)alloc((size_t)HK * NB * NW * DH * 4);
  float* cvbuf = (float*)alloc((size_t)HK * NB * NW * DH * 4);
  float* ckf   = (float*)alloc((size_t)NB * HK * 16 * DH * 4);
  float* cvf   = (float*)alloc((size_t)NB * HK * 16 * DH * 4);
  unsigned* selm = (unsigned*)alloc((size_t)NB * HK * BALL * 4);
  float* coutb = (float*)alloc((size_t)NTOK * H * DH * 4);
  float* foutb = (float*)alloc((size_t)NTOK * H * DH * 4);
  float* soutb = (float*)alloc((size_t)NTOK * H * DH * 4);
  bf16*  aoutb = (bf16*) alloc((size_t)NTOK * H * DH * 2);
  // transposed bf16 weights: Bt[N][K]
  bf16*  wqb   = (bf16*) alloc((size_t)DIM * H * DH * 2);
  bf16*  wkb   = (bf16*) alloc((size_t)DIM * HK * DH * 2);
  bf16*  wvb   = (bf16*) alloc((size_t)DIM * HK * DH * 2);
  bf16*  wgb   = (bf16*) alloc((size_t)DIM * 3 * H * 2);
  bf16*  wck1b = (bf16*) alloc((size_t)HK * WD * WD * 2);
  bf16*  wck2b = (bf16*) alloc((size_t)HK * WD * DH * 2);
  bf16*  wcv1b = (bf16*) alloc((size_t)HK * WD * WD * 2);
  bf16*  wcv2b = (bf16*) alloc((size_t)HK * WD * DH * 2);
  bf16*  wob   = (bf16*) alloc((size_t)H * DH * DIM * 2);

  auto cvtT = [&](const float* s, bf16* d, int K, int N) {
    size_t n = (size_t)K * N;
    f2bf_t<<<dim3((unsigned)((n + 255) / 256)), 256, 0, stream>>>(s, d, K, N);
  };
  cvtT(Wq, wqb, DIM, H * DH);
  cvtT(Wk, wkb, DIM, HK * DH);
  cvtT(Wv, wvb, DIM, HK * DH);
  cvtT(Wg, wgb, DIM, 3 * H);
  for (int hk = 0; hk < HK; ++hk) {
    cvtT(Wck1 + (size_t)hk * WD * WD, wck1b + (size_t)hk * WD * WD, WD, WD);
    cvtT(Wck2 + (size_t)hk * WD * DH, wck2b + (size_t)hk * WD * DH, WD, DH);
    cvtT(Wcv1 + (size_t)hk * WD * WD, wcv1b + (size_t)hk * WD * WD, WD, WD);
    cvtT(Wcv2 + (size_t)hk * WD * DH, wcv2b + (size_t)hk * WD * DH, WD, DH);
  }
  cvtT(Wo, wob, H * DH, DIM);

  prep<<<NB, 256, 0, stream>>>(x, pos, wpe, bpe, gamma, xn, xnb);

  auto gemm = [&](const bf16* A, const bf16* Bt, float* C, bf16* Cb, int M, int N,
                  int K, int act) {
    gemm_bf16<<<dim3((N + 63) / 64, (M + 127) / 128), 256, 0, stream>>>(
        A, Bt, C, Cb, M, N, K, act);
  };
  gemm(xnb, wqb, qf, nullptr, NTOK, H * DH, DIM, 0);
  gemm(xnb, wkb, kf, nullptr, NTOK, HK * DH, DIM, 0);
  gemm(xnb, wvb, vf, nullptr, NTOK, HK * DH, DIM, 0);
  gemm(xnb, wgb, graw, nullptr, NTOK, 3 * H, DIM, 0);

  rope<<<NTOK, 512, 0, stream>>>(qf, kf, vf, qr, qrb, krb, vtb);

  {
    size_t tot = (size_t)HK * NB * NW * WD;
    windows<<<dim3((unsigned)((tot + 255) / 256)), 256, 0, stream>>>(kf, vf, kpos,
                                                                     vpos, kw, vw);
  }

  for (int hk = 0; hk < HK; ++hk) {
    gemm(kw + (size_t)hk * NB * NW * WD, wck1b + (size_t)hk * WD * WD, nullptr,
         hid, NB * NW, WD, WD, 1);
    gemm(hid, wck2b + (size_t)hk * WD * DH, ckbuf + (size_t)hk * NB * NW * DH,
         nullptr, NB * NW, DH, WD, 0);
  }
  for (int hk = 0; hk < HK; ++hk) {
    gemm(vw + (size_t)hk * NB * NW * WD, wcv1b + (size_t)hk * WD * WD, nullptr,
         hid, NB * NW, WD, WD, 1);
    gemm(hid, wcv2b + (size_t)hk * WD * DH, cvbuf + (size_t)hk * NB * NW * DH,
         nullptr, NB * NW, DH, WD, 0);
  }

  assemble<<<(NB * HK * 16 * DH) / 256, 256, 0, stream>>>(ckbuf, cvbuf, memk, memv,
                                                          ckf, cvf);
  cmp_attn<<<NB * HK, 256, 0, stream>>>(qr, ckf, cvf, coutb, selm);
  fine_attn<<<dim3(BALL / 32, G, NB * HK), 256, 0, stream>>>(qrb, krb, vtb, selm,
                                                             foutb, soutb);
  {
    size_t tot = (size_t)NTOK * H * DH;
    combine<<<dim3((unsigned)(tot / 256)), 256, 0, stream>>>(graw, coutb, foutb,
                                                             soutb, aoutb);
  }
  gemm(aoutb, wob, out, nullptr, NTOK, DIM, H * DH, 0);
}
